// MedAdapter_67997922230947
// MI455X (gfx1250) — matrix operand order repaired and checked by reading, not compiled, not run on hardware
//
#include <hip/hip_runtime.h>

// ---------------------------------------------------------------------------
// MedAdapter block for MI455X (gfx1250, wave32, WMMA bf16)
// B=2 T=32 H=32 W=32, IN=384 HID=64 OUT=384, rows = B*T*H*W = 65536
// ---------------------------------------------------------------------------

typedef __attribute__((ext_vector_type(16))) __bf16 v16bf;
typedef __attribute__((ext_vector_type(8)))  float  v8f;
typedef __attribute__((ext_vector_type(2)))  float  v2f;
typedef __attribute__((ext_vector_type(2)))  __bf16 v2bf;

union BfFrag { unsigned int u[8]; v16bf v; };

// native packed f32 -> bf16 conversion (backend picks v_cvt_pk_bf16_f32)
__device__ __forceinline__ unsigned int cvt2(v2f f) {
  v2bf b = __builtin_convertvector(f, v2bf);
  return __builtin_bit_cast(unsigned int, b);
}
__device__ __forceinline__ float gelu_exact(float v) {
  return 0.5f * v * (1.0f + erff(v * 0.70710678118654752f));
}
__device__ __forceinline__ v8f wmma_bf16(const BfFrag& a, const BfFrag& b, v8f c) {
  return __builtin_amdgcn_wmma_f32_16x16x32_bf16(
      false, a.v, false, b.v, (short)0, c, false, false);
}
// A-fragment K index for VGPR v (16-bit A 16x32): interleave split at v=4, half offset 8
__device__ __forceinline__ int kA(int v, int half) {
  return ((v < 4) ? 2 * v : 16 + 2 * (v - 4)) + half * 8;
}
// B-fragment K index: same interleave as A (split at v=4, half offset 8)
__device__ __forceinline__ int kB(int v, int half) { return ((v < 4) ? 2 * v : 16 + 2 * (v - 4)) + half * 8; }

// ---------------------------------------------------------------------------
// 0a) one-shot f32 -> bf16 weight conversion (shared by all GEMM blocks)
// ---------------------------------------------------------------------------
__global__ void k_cvtw(const float* __restrict__ dw, const float* __restrict__ uw,
                       const float* __restrict__ cw,
                       __bf16* __restrict__ dwb, __bf16* __restrict__ uwb,
                       __bf16* __restrict__ cwb) {
  const int i = blockIdx.x * 256 + threadIdx.x;
  if (i < 24576) { dwb[i] = (__bf16)dw[i]; uwb[i] = (__bf16)uw[i]; }
  if (i < 4096)  { cwb[i] = (__bf16)cw[i]; }
}

// ---------------------------------------------------------------------------
// 0b) prep: kw = ifft(fw), d = ifft(fb)  (32-point IDFT, numpy sign e^{+i2pikn/N}/N)
// kd layout: kwre[32] kwim[32] dre[32] dim[32]
// ---------------------------------------------------------------------------
__global__ void k_prep(const float* __restrict__ fw, const float* __restrict__ fb,
                       float* __restrict__ kd) {
  int t = threadIdx.x;
  if (t < 32) {
    float kr = 0.f, ki = 0.f, dr = 0.f, di = 0.f;
    for (int k = 0; k < 32; ++k) {
      float ang = 6.283185307179586f * (float)((k * t) & 31) * (1.0f / 32.0f);
      float cs = cosf(ang), sn = sinf(ang);
      kr += fw[k] * cs; ki += fw[k] * sn;
      dr += fb[k] * cs; di += fb[k] * sn;
    }
    const float s = 1.0f / 32.0f;
    kd[t] = kr * s; kd[32 + t] = ki * s; kd[64 + t] = dr * s; kd[96 + t] = di * s;
  }
}

// ---------------------------------------------------------------------------
// 1) down GEMM + GELU:  h5[b,c,t,h,w] = gelu(x[row,:] @ down_w.T + down_b)
//    M=65536 N=64 K=384, bf16 WMMA, f32 accum.  1 block = 64 rows x 64 cols.
// ---------------------------------------------------------------------------
__global__ void __launch_bounds__(128)
k_down_gemm(const float* __restrict__ x, const __bf16* __restrict__ dwb,
            const float* __restrict__ db, float* __restrict__ h5) {
  __shared__ float hT[64 * 65];                 // transpose staging (65: bank-safe)
  const int tid = threadIdx.x, wave = tid >> 5, lane = tid & 31;
  const int m = lane & 15, half = lane >> 4;
  const int rowbase = blockIdx.x * 64;
  const int row = rowbase + wave * 16 + m;

  v8f acc[4] = {};
#pragma unroll 2
  for (int kt = 0; kt < 12; ++kt) {
    const int kbase = kt * 32;
    BfFrag A;
#pragma unroll
    for (int v = 0; v < 8; ++v) {
      int kk = kbase + kA(v, half);
      A.u[v] = cvt2(*(const v2f*)&x[(size_t)row * 384 + kk]);
    }
#pragma unroll
    for (int nt = 0; nt < 4; ++nt) {
      BfFrag Bf;
      const int n = nt * 16 + m;                // B[k][n] = down_w[n][k]
#pragma unroll
      for (int v = 0; v < 8; ++v) {
        int kk = kbase + kB(v, half);
        Bf.u[v] = *(const unsigned int*)&dwb[n * 384 + kk];
      }
      acc[nt] = wmma_bf16(A, Bf, acc[nt]);
    }
  }
  // epilogue: bias + exact GELU, transpose through LDS so h5 writes coalesce
#pragma unroll
  for (int nt = 0; nt < 4; ++nt) {
    const int n = nt * 16 + m;
    const float bias = db[n];
#pragma unroll
    for (int r = 0; r < 8; ++r) {
      int mr = wave * 16 + half * 8 + r;        // D: M = r + 8*half
      hT[mr * 65 + n] = gelu_exact(acc[nt][r] + bias);
    }
  }
  __syncthreads();
  const int bi = rowbase >> 15;                 // batch
  const int pbase = rowbase & 32767;            // t*1024+h*32+w base
  for (int i = 0; i < 32; ++i) {
    int e = tid + i * 128;
    int rl = e & 63, c = e >> 6;
    h5[((size_t)(bi * 64 + c)) * 32768 + pbase + rl] = hT[rl * 65 + c];
  }
}

// ---------------------------------------------------------------------------
// 2) fused spatial depthwise convs (1,3,3) and (1,5,5), halo tile in LDS
// ---------------------------------------------------------------------------
__global__ void __launch_bounds__(256)
k_spatial(const float* __restrict__ h5,
          const float* __restrict__ k31w, const float* __restrict__ k31b,
          const float* __restrict__ k51w, const float* __restrict__ k51b,
          float* __restrict__ t1, float* __restrict__ t2) {
  __shared__ float sh[36 * 36];
  const int tid = threadIdx.x;
  const int bc = blockIdx.x >> 5, t = blockIdx.x & 31, c = bc & 63;
  const float* plane = h5 + (size_t)bc * 32768 + t * 1024;
  for (int e = tid; e < 1296; e += 256) {
    int y = e / 36 - 2, xq = e % 36 - 2;
    float v = 0.f;
    if ((unsigned)y < 32u && (unsigned)xq < 32u) v = plane[y * 32 + xq];
    sh[e] = v;
  }
  __syncthreads();
  float w3[9], w5[25];
#pragma unroll
  for (int i = 0; i < 9; ++i)  w3[i] = k31w[c * 9 + i];
#pragma unroll
  for (int i = 0; i < 25; ++i) w5[i] = k51w[c * 25 + i];
  const float b3 = k31b[c], b5 = k51b[c];
  for (int j = 0; j < 4; ++j) {
    const int p = tid + j * 256;
    const int hy = p >> 5, wx = p & 31;
    const float* cp = &sh[(hy + 2) * 36 + (wx + 2)];
    float a3 = b3, a5 = b5;
#pragma unroll
    for (int dy = -1; dy <= 1; ++dy)
#pragma unroll
      for (int dx = -1; dx <= 1; ++dx)
        a3 += cp[dy * 36 + dx] * w3[(dy + 1) * 3 + dx + 1];
#pragma unroll
    for (int dy = -2; dy <= 2; ++dy)
#pragma unroll
      for (int dx = -2; dx <= 2; ++dx)
        a5 += cp[dy * 36 + dx] * w5[(dy + 2) * 5 + dx + 2];
    const size_t oi = (size_t)bc * 32768 + t * 1024 + p;
    t1[oi] = a3; t2[oi] = a5;
  }
}

// ---------------------------------------------------------------------------
// 3) fused temporal depthwise convs (3,1,1)/(5,1,1) + BN sum/sumsq reduction
//    sums: s1[64] q1[64] s2[64] q2[64], accumulated with global f32 atomics
// ---------------------------------------------------------------------------
__global__ void __launch_bounds__(256)
k_temporal(const float* __restrict__ t1, const float* __restrict__ t2,
           const float* __restrict__ k32w, const float* __restrict__ k32b,
           const float* __restrict__ k52w, const float* __restrict__ k52b,
           float* __restrict__ x1pre, float* __restrict__ x2pre,
           float* __restrict__ sums) {
  const int tid = threadIdx.x;
  const int bc = blockIdx.x, c = bc & 63;
  const float* p1 = t1 + (size_t)bc * 32768;
  const float* p2 = t2 + (size_t)bc * 32768;
  float w3[3], w5[5];
#pragma unroll
  for (int i = 0; i < 3; ++i) w3[i] = k32w[c * 3 + i];
#pragma unroll
  for (int i = 0; i < 5; ++i) w5[i] = k52w[c * 5 + i];
  const float b3 = k32b[c], b5 = k52b[c];
  float s1 = 0.f, q1 = 0.f, s2 = 0.f, q2 = 0.f;
  for (int i = 0; i < 128; ++i) {
    const int e = tid + i * 256;
    const int t = e >> 10, hw = e & 1023;
    float a1 = b3, a2 = b5;
#pragma unroll
    for (int d = 0; d < 3; ++d) { int tt = t + d - 1; if ((unsigned)tt < 32u) a1 += p1[tt * 1024 + hw] * w3[d]; }
#pragma unroll
    for (int d = 0; d < 5; ++d) { int tt = t + d - 2; if ((unsigned)tt < 32u) a2 += p2[tt * 1024 + hw] * w5[d]; }
    x1pre[(size_t)bc * 32768 + e] = a1;
    x2pre[(size_t)bc * 32768 + e] = a2;
    s1 += a1; q1 += a1 * a1; s2 += a2; q2 += a2 * a2;
  }
  __shared__ float red[256 * 4];
  red[tid] = s1; red[256 + tid] = q1; red[512 + tid] = s2; red[768 + tid] = q2;
  __syncthreads();
  for (int st = 128; st > 0; st >>= 1) {
    if (tid < st) {
      red[tid] += red[tid + st];             red[256 + tid] += red[256 + tid + st];
      red[512 + tid] += red[512 + tid + st]; red[768 + tid] += red[768 + tid + st];
    }
    __syncthreads();
  }
  if (tid == 0) {
    atomicAdd(&sums[c],        red[0]);
    atomicAdd(&sums[64 + c],   red[256]);
    atomicAdd(&sums[128 + c],  red[512]);
    atomicAdd(&sums[192 + c],  red[768]);
  }
}

// ---------------------------------------------------------------------------
// 4) fold BN stats into scale/shift:  y = x*scale + shift  (then ReLU later)
// ---------------------------------------------------------------------------
__global__ void k_finalize(const float* __restrict__ sums,
                           const float* __restrict__ g3, const float* __restrict__ b3,
                           const float* __restrict__ g5, const float* __restrict__ b5,
                           float* __restrict__ sc) {
  const int c = threadIdx.x;
  if (c < 64) {
    const float inv_n = 1.0f / 65536.0f;
    float m1 = sums[c] * inv_n;
    float v1 = sums[64 + c] * inv_n - m1 * m1;
    float s1 = rsqrtf(v1 + 1e-5f) * g3[c];
    sc[c] = s1; sc[64 + c] = b3[c] - m1 * s1;
    float m2 = sums[128 + c] * inv_n;
    float v2 = sums[192 + c] * inv_n - m2 * m2;
    float s2 = rsqrtf(v2 + 1e-5f) * g5[c];
    sc[128 + c] = s2; sc[192 + c] = b5[c] - m2 * s2;
  }
}

// ---------------------------------------------------------------------------
// 5) frequency branch, algebraically reduced: since fw/fb vary only along the
//    T-frequency axis, ifftn(fftn(h)*fw + fb) = circconv_T(h, ifft(fw))
//    + ifft(fb)[t] * delta(h==0 && w==0).  x3 = |...|.
// ---------------------------------------------------------------------------
__global__ void __launch_bounds__(256)
k_freq(const float* __restrict__ h5, const float* __restrict__ kd,
       float* __restrict__ x3) {
  __shared__ float kwre[32], kwim[32], dre[32], dim_[32];
  const int tid = threadIdx.x;
  if (tid < 32) { kwre[tid] = kd[tid]; kwim[tid] = kd[32 + tid];
                  dre[tid] = kd[64 + tid]; dim_[tid] = kd[96 + tid]; }
  __syncthreads();
  const int bc = blockIdx.x >> 2;
  const int hw = (blockIdx.x & 3) * 256 + tid;
  const float* base = h5 + (size_t)bc * 32768 + hw;
  float hv[32];
#pragma unroll
  for (int t = 0; t < 32; ++t) hv[t] = base[t * 1024];
  const bool origin = (hw == 0);
  float* ob = x3 + (size_t)bc * 32768 + hw;
  for (int t = 0; t < 32; ++t) {
    float re = origin ? dre[t] : 0.f;
    float im = origin ? dim_[t] : 0.f;
#pragma unroll
    for (int src = 0; src < 32; ++src) {        // hv[] stays in registers
      const int tau = (t - src) & 31;
      re += kwre[tau] * hv[src];
      im += kwim[tau] * hv[src];
    }
    ob[t * 1024] = sqrtf(re * re + im * im);
  }
}

// ---------------------------------------------------------------------------
// 6) BN+ReLU + branch sum + 64x64 pointwise conv (WMMA) + residual.
//    Writes y directly in [row=b*32768+p][c] bf16 layout = up-GEMM A matrix.
// ---------------------------------------------------------------------------
__global__ void __launch_bounds__(128)
k_bn_pw(const float* __restrict__ x1pre, const float* __restrict__ x2pre,
        const float* __restrict__ x3, const float* __restrict__ h5,
        const __bf16* __restrict__ cwb, const float* __restrict__ cb,
        const float* __restrict__ sc, __bf16* __restrict__ ypos) {
  __shared__ __align__(16) __bf16 sT[64 * 66];           // s tile, bf16, stride 66
  __shared__ float yT[64 * 65];                          // y tile, f32, stride 65
  const int tid = threadIdx.x, wave = tid >> 5, lane = tid & 31;
  const int m = lane & 15, half = lane >> 4;
  const int bi = blockIdx.x >> 9;
  const int pbase = (blockIdx.x & 511) * 64;
  const float* scale1 = sc;       const float* shift1 = sc + 64;
  const float* scale2 = sc + 128; const float* shift2 = sc + 192;

  // s = relu(bn3(x1pre)) + relu(bn5(x2pre)) + x3   -> LDS as bf16 [n][c]
  for (int i = 0; i < 32; ++i) {
    const int e = tid + i * 128;
    const int n = e & 63, c = e >> 6;
    const size_t idx = ((size_t)(bi * 64 + c)) * 32768 + pbase + n;
    float v1 = x1pre[idx] * scale1[c] + shift1[c]; v1 = v1 > 0.f ? v1 : 0.f;
    float v2 = x2pre[idx] * scale2[c] + shift2[c]; v2 = v2 > 0.f ? v2 : 0.f;
    sT[n * 66 + c] = (__bf16)(v1 + v2 + x3[idx]);
  }
  __syncthreads();

  // attn = conv_w @ s : M=64(o, 1 o-tile per wave) N=64(pos) K=64(c)
  const int ot = wave;
  v8f acc[4] = {};
#pragma unroll
  for (int kt = 0; kt < 2; ++kt) {
    const int kbase = kt * 32;
    BfFrag A;
    const int o = ot * 16 + m;
#pragma unroll
    for (int v = 0; v < 8; ++v) {
      int kk = kbase + kA(v, half);
      A.u[v] = *(const unsigned int*)&cwb[o * 64 + kk];
    }
#pragma unroll
    for (int nt = 0; nt < 4; ++nt) {
      BfFrag Bf;
      const int n = nt * 16 + m;
#pragma unroll
      for (int v = 0; v < 8; ++v) {
        int kk = kbase + kB(v, half);
        Bf.u[v] = *(const unsigned int*)&sT[n * 66 + kk];  // 4B-aligned pairs
      }
      acc[nt] = wmma_bf16(A, Bf, acc[nt]);
    }
  }
  // y = h5 + attn + conv_b  (h5 reads coalesced in N), stage to LDS
#pragma unroll
  for (int nt = 0; nt < 4; ++nt) {
    const int n = nt * 16 + m;
#pragma unroll
    for (int r = 0; r < 8; ++r) {
      const int oo = ot * 16 + half * 8 + r;
      const int p = pbase + n;
      float val = acc[nt][r] + cb[oo] + h5[((size_t)(bi * 64 + oo)) * 32768 + p];
      yT[n * 65 + oo] = val;
    }
  }
  __syncthreads();
  for (int i = 0; i < 32; ++i) {
    const int e = tid + i * 128;
    const int o = e & 63, n = e >> 6;
    const size_t row = (size_t)bi * 32768 + pbase + n;
    ypos[row * 64 + o] = (__bf16)yT[n * 65 + o];    // contiguous 128B per row
  }
}

// ---------------------------------------------------------------------------
// 7) up GEMM + GELU + residual: out = x + gelu(ypos @ up_w.T + up_b)
//    M=65536 N=384 K=64; A already bf16.
// ---------------------------------------------------------------------------
__global__ void __launch_bounds__(128)
k_up_gemm(const float* __restrict__ x, const __bf16* __restrict__ ypos,
          const __bf16* __restrict__ uwb, const float* __restrict__ ub,
          float* __restrict__ out) {
  const int tid = threadIdx.x, wave = tid >> 5, lane = tid & 31;
  const int m = lane & 15, half = lane >> 4;
  const int rowbase = blockIdx.x * 64;
  const int colbase = blockIdx.y * 64;
  const int row = rowbase + wave * 16 + m;

  v8f acc[4] = {};
#pragma unroll
  for (int kt = 0; kt < 2; ++kt) {
    const int kbase = kt * 32;
    BfFrag A;
#pragma unroll
    for (int v = 0; v < 8; ++v) {
      int kk = kbase + kA(v, half);
      A.u[v] = *(const unsigned int*)&ypos[(size_t)row * 64 + kk];
    }
#pragma unroll
    for (int nt = 0; nt < 4; ++nt) {
      BfFrag Bf;
      const int n = colbase + nt * 16 + m;       // B[k][n] = up_w[n][k]
#pragma unroll
      for (int v = 0; v < 8; ++v) {
        int kk = kbase + kB(v, half);
        Bf.u[v] = *(const unsigned int*)&uwb[n * 64 + kk];
      }
      acc[nt] = wmma_bf16(A, Bf, acc[nt]);
    }
  }
#pragma unroll
  for (int nt = 0; nt < 4; ++nt) {
    const int col = colbase + nt * 16 + m;
    const float bias = ub[col];
#pragma unroll
    for (int r = 0; r < 8; ++r) {
      const int rw = rowbase + wave * 16 + half * 8 + r;
      const size_t o = (size_t)rw * 384 + col;
      out[o] = x[o] + gelu_exact(acc[nt][r] + bias);
    }
  }
}

// ---------------------------------------------------------------------------
// launch
// ---------------------------------------------------------------------------
extern "C" void kernel_launch(void* const* d_in, const int* in_sizes, int n_in,
                              void* d_out, int out_size, void* d_ws, size_t ws_size,
                              hipStream_t stream) {
  const float* x      = (const float*)d_in[0];
  const float* down_w = (const float*)d_in[1];
  const float* down_b = (const float*)d_in[2];
  const float* k31w   = (const float*)d_in[3];
  const float* k31b   = (const float*)d_in[4];
  const float* k32w   = (const float*)d_in[5];
  const float* k32b   = (const float*)d_in[6];
  const float* bn3g   = (const float*)d_in[7];
  const float* bn3b   = (const float*)d_in[8];
  const float* k51w   = (const float*)d_in[9];
  const float* k51b   = (const float*)d_in[10];
  const float* k52w   = (const float*)d_in[11];
  const float* k52b   = (const float*)d_in[12];
  const float* bn5g   = (const float*)d_in[13];
  const float* bn5b   = (const float*)d_in[14];
  const float* conv_w = (const float*)d_in[15];
  const float* conv_b = (const float*)d_in[16];
  const float* fw     = (const float*)d_in[17];
  const float* fb     = (const float*)d_in[18];
  const float* up_w   = (const float*)d_in[19];
  const float* up_b   = (const float*)d_in[20];
  float* out = (float*)d_out;

  // workspace layout (f32 elements unless noted)
  const size_t VOL = 2ull * 64ull * 32768ull;  // 4,194,304
  float* W     = (float*)d_ws;
  float* h5    = W;
  float* t1    = h5 + VOL;
  float* t2    = t1 + VOL;
  float* x1pre = t2 + VOL;
  float* x2pre = x1pre + VOL;
  float* x3    = x2pre + VOL;
  float* sums  = x3 + VOL;          // 256 f32
  float* sc    = sums + 256;        // 256 f32
  float* kd    = sc + 256;          // 128 f32
  __bf16* ypos = (__bf16*)(kd + 128);   // 65536*64 bf16
  __bf16* dwb  = ypos + VOL;            // 24576 bf16
  __bf16* uwb  = dwb + 24576;           // 24576 bf16
  __bf16* cwb  = uwb + 24576;           // 4096 bf16

  (void)hipMemsetAsync(sums, 0, 256 * sizeof(float), stream);
  k_cvtw<<<96, 256, 0, stream>>>(down_w, up_w, conv_w, dwb, uwb, cwb);
  k_prep<<<1, 32, 0, stream>>>(fw, fb, kd);
  k_down_gemm<<<1024, 128, 0, stream>>>(x, dwb, down_b, h5);
  k_spatial<<<4096, 256, 0, stream>>>(h5, k31w, k31b, k51w, k51b, t1, t2);
  k_temporal<<<128, 256, 0, stream>>>(t1, t2, k32w, k32b, k52w, k52b,
                                      x1pre, x2pre, sums);
  k_finalize<<<1, 64, 0, stream>>>(sums, bn3g, bn3b, bn5g, bn5b, sc);
  k_freq<<<512, 256, 0, stream>>>(h5, kd, x3);
  k_bn_pw<<<1024, 128, 0, stream>>>(x1pre, x2pre, x3, h5, cwb, conv_b, sc, ypos);
  k_up_gemm<<<dim3(1024, 6), 128, 0, stream>>>(x, ypos, uwb, up_b, out);
}
